// LinearAttention_3539053052596
// MI455X (gfx1250) — matrix operand order repaired and checked by reading, not compiled, not run on hardware
//
#include <hip/hip_runtime.h>
#include <hip/hip_bf16.h>
#include <cstdint>
#include <cstddef>

// ---------------------------------------------------------------------------
// LinearAttention for MI455X (gfx1250, wave32, WMMA).
// Pipeline:
//   K0: convert W_qkv / W_out f32 -> bf16 (once, tiny)
//   K1: qkv GEMM  (W_qkv_bf16 @ x) -> Q^T bf16 [b][l][c'], K f32 [b][dk][l],
//       V bf16 [b][dv][l].  bf16 WMMA, LDS-transposed x tiles (packed b32).
//   K2: row softmax on K (l=4096) -> K_sm bf16
//   K3: context = K_sm @ V^T per (b,h) -> ctx bf16; double-buffered
//       GLOBAL_LOAD_ASYNC_TO_LDS_B128 staging, s_wait_asynccnt throttling.
//   K4: fold W_out with ctx: M[b][o][h*64+d] = sum_e Wout[o][h*64+e]*ctx[d][e]
//   K5: final = M @ Q + b_out -> d_out f32  (bf16 WMMA, no LDS: Q^T layout)
// ---------------------------------------------------------------------------

typedef __bf16 bf16_t;
typedef bf16_t v16bf __attribute__((ext_vector_type(16)));
typedef bf16_t v8bf  __attribute__((ext_vector_type(8)));
typedef float  v8f   __attribute__((ext_vector_type(8)));

union FragU { v16bf v; v8bf h[2]; };

// Problem constants
#define BATCH 16
#define CIN   256
#define LDIM  4096
#define HID   512
#define HEADS 8
#define DHEAD 64
#define OQKV  1536   // 3*HID

// Workspace carve (bytes, all 256-aligned)
#define OFF_WQKV 0ull                         // 1536*256*2      =    786432
#define OFF_WOUT 786432ull                    // 256*512*2       =    262144
#define OFF_QT   1048576ull                   // 16*4096*512*2   =  67108864
#define OFF_K    68157440ull                  // 16*512*4096*4   = 134217728
#define OFF_KSM  202375168ull                 // 16*512*4096*2   =  67108864
#define OFF_V    269484032ull                 // 16*512*4096*2   =  67108864
#define OFF_CTX  336592896ull                 // 16*8*64*64*2    =   1048576
#define OFF_M    337641472ull                 // 16*256*512*2    =   4194304

__device__ __forceinline__ v8f wmma_bf16(v16bf a, v16bf b, v8f c) {
  // D = A(16x32) * B(32x16) + C, f32 accumulate
  return __builtin_amdgcn_wmma_f32_16x16x32_bf16(
      false, a, false, b, (short)0, c, false, false);
}

// A fragment (16x32 bf16, ISA 7.12.2): lane m=lane&15, half=lane>>4
// a[0..7] = A[m][8*half+0..7], a[8..15] = A[m][8*half+16..23]
__device__ __forceinline__ v16bf frag_a(const bf16_t* __restrict__ base, int ld) {
  const int lane = threadIdx.x & 31;
  const int m = lane & 15, half = lane >> 4;
  const bf16_t* p = base + (size_t)m * ld + 8 * half;
  FragU f;
  f.h[0] = *(const v8bf*)(p);
  f.h[1] = *(const v8bf*)(p + 16);
  return f.v;
}

// B fragment (32x16 bf16) from B^T rows (row n holds K-contiguous data):
// b[0..7] = Bt[n][8*half+0..7], b[8..15] = Bt[n][8*half+16..23]
__device__ __forceinline__ v16bf frag_b(const bf16_t* __restrict__ baseT, int ld) {
  const int lane = threadIdx.x & 31;
  const int n = lane & 15, half = lane >> 4;
  const bf16_t* p = baseT + (size_t)n * ld + 8 * half;
  FragU f;
  f.h[0] = *(const v8bf*)(p);
  f.h[1] = *(const v8bf*)(p + 16);
  return f.v;
}

// 64B global -> LDS async DMA (4x b128), tracked by ASYNCcnt.
__device__ __forceinline__ void async_copy64B(uint32_t ldsaddr, const bf16_t* src) {
  asm volatile(
      "global_load_async_to_lds_b128 %0, %1, off\n\t"
      "global_load_async_to_lds_b128 %0, %1, off offset:16\n\t"
      "global_load_async_to_lds_b128 %0, %1, off offset:32\n\t"
      "global_load_async_to_lds_b128 %0, %1, off offset:48"
      :: "v"(ldsaddr), "v"(src) : "memory");
}

// ---------------------------------------------------------------------------
// K0: convert weights to bf16 (wqkv: 393216 elems, wout: 131072 elems)
__global__ __launch_bounds__(256) void k_convert_weights(
    const float* __restrict__ wqkv, const float* __restrict__ wout,
    bf16_t* __restrict__ wqkv_bf, bf16_t* __restrict__ wout_bf) {
  const int i = blockIdx.x * 256 + threadIdx.x;
  if (i < OQKV * CIN) wqkv_bf[i] = (bf16_t)wqkv[i];
  if (i < CIN * HID)  wout_bf[i] = (bf16_t)wout[i];
}

// ---------------------------------------------------------------------------
// K1: qkv GEMM. grid (64 l-tiles, 24 o-tiles, 16 b), 128 threads (4 waves).
// Each block: 64(o) x 64(l) tile, K-loop over c=256 in steps of 32.
#define XLDB 40   // padded LDS row stride (elems, even) to dodge bank conflicts
__global__ __launch_bounds__(128) void k_qkv_gemm(
    const float* __restrict__ x, const bf16_t* __restrict__ wq,
    bf16_t* __restrict__ qt, float* __restrict__ kbuf,
    bf16_t* __restrict__ vbuf) {
  __shared__ bf16_t xt[64 * XLDB];   // x tile transposed: [l][k]

  const int b   = blockIdx.z;
  const int ot  = blockIdx.y;        // 0..23
  const int l0  = blockIdx.x * 64;
  const int og0 = ot * 64;
  const int wv  = threadIdx.x >> 5;
  const int lane = threadIdx.x & 31;
  const int mq = wv >> 1, nq = wv & 1;

  v8f acc[2][2] = {};
  const size_t xbase = (size_t)b * CIN * LDIM;

  for (int k0 = 0; k0 < CIN; k0 += 32) {
    // stage x tile (32 k x 64 l) -> LDS transposed bf16 [l][k].
    // Thread owns k-row pair {2kp,2kp+1} x 8 l: pack 2 bf16 per dword store.
    {
      const int kp = threadIdx.x >> 3;         // 0..15
      const int lc = (threadIdx.x & 7) * 8;    // 0..56
      const float* xp0 = x + xbase + (size_t)(k0 + 2 * kp) * LDIM + l0 + lc;
      const float* xp1 = xp0 + LDIM;
      const float4* q0 = (const float4*)xp0;
      const float4* q1 = (const float4*)xp1;
      float4 a0 = q0[0], a1 = q0[1];
      float4 b0 = q1[0], b1 = q1[1];
      if (k0 + 32 < CIN) __builtin_prefetch(xp0 + 32 * LDIM, 0, 0);
      float r0[8] = {a0.x, a0.y, a0.z, a0.w, a1.x, a1.y, a1.z, a1.w};
      float r1[8] = {b0.x, b0.y, b0.z, b0.w, b1.x, b1.y, b1.z, b1.w};
      uint32_t* xtw = (uint32_t*)xt;
#pragma unroll
      for (int i = 0; i < 8; i++) {
        union { bf16_t h[2]; uint32_t u; } w;
        w.h[0] = (bf16_t)r0[i];
        w.h[1] = (bf16_t)r1[i];
        xtw[((lc + i) * XLDB + 2 * kp) >> 1] = w.u;
      }
    }
    __syncthreads();

    v16bf af[2], bfr[2];
#pragma unroll
    for (int mi = 0; mi < 2; mi++)
      af[mi] = frag_a(wq + (size_t)(og0 + mq * 32 + mi * 16) * CIN + k0, CIN);
#pragma unroll
    for (int ni = 0; ni < 2; ni++)
      bfr[ni] = frag_b(&xt[(nq * 32 + ni * 16) * XLDB], XLDB);
#pragma unroll
    for (int mi = 0; mi < 2; mi++)
#pragma unroll
      for (int ni = 0; ni < 2; ni++)
        acc[mi][ni] = wmma_bf16(af[mi], bfr[ni], acc[mi][ni]);
    __syncthreads();
  }

  const int n = lane & 15, half = lane >> 4;
  if (ot < 8) {
    // Q region: store TRANSPOSED bf16 [b][l][oq]; per lane one packed 16B store
#pragma unroll
    for (int mi = 0; mi < 2; mi++)
#pragma unroll
      for (int ni = 0; ni < 2; ni++) {
        const int oq = og0 + mq * 32 + mi * 16 + 8 * half;
        const int l  = l0 + nq * 32 + ni * 16 + n;
        v8bf pk;
#pragma unroll
        for (int j = 0; j < 8; j++) pk[j] = (bf16_t)acc[mi][ni][j];
        *(v8bf*)(qt + ((size_t)b * LDIM + l) * HID + oq) = pk;
      }
  } else if (ot < 16) {
    // K region: keep f32 for softmax, layout [b][dk][l]
#pragma unroll
    for (int mi = 0; mi < 2; mi++)
#pragma unroll
      for (int ni = 0; ni < 2; ni++) {
        const int dk = og0 - HID + mq * 32 + mi * 16 + 8 * half;
        const int l  = l0 + nq * 32 + ni * 16 + n;
#pragma unroll
        for (int j = 0; j < 8; j++)
          kbuf[((size_t)b * HID + dk + j) * LDIM + l] = acc[mi][ni][j];
      }
  } else {
    // V region: bf16, layout [b][dv][l]
#pragma unroll
    for (int mi = 0; mi < 2; mi++)
#pragma unroll
      for (int ni = 0; ni < 2; ni++) {
        const int dv = og0 - 2 * HID + mq * 32 + mi * 16 + 8 * half;
        const int l  = l0 + nq * 32 + ni * 16 + n;
#pragma unroll
        for (int j = 0; j < 8; j++)
          vbuf[((size_t)b * HID + dv + j) * LDIM + l] = (bf16_t)acc[mi][ni][j];
      }
  }
}

// ---------------------------------------------------------------------------
// K2: softmax over l for each K row. grid 16*512 blocks, 256 threads.
__global__ __launch_bounds__(256) void k_softmax(
    const float* __restrict__ kin, bf16_t* __restrict__ ksm) {
  __shared__ float red[256];
  const size_t row = blockIdx.x;
  const float* p = kin + row * LDIM;
  bf16_t* o = ksm + row * LDIM;
  const int t = threadIdx.x;

  float xv[16];
  const float4* p4 = (const float4*)p;
#pragma unroll
  for (int i = 0; i < 4; i++) {
    float4 f = p4[t * 4 + i];
    xv[4 * i + 0] = f.x; xv[4 * i + 1] = f.y;
    xv[4 * i + 2] = f.z; xv[4 * i + 3] = f.w;
  }
  float mx = xv[0];
#pragma unroll
  for (int i = 1; i < 16; i++) mx = fmaxf(mx, xv[i]);
  red[t] = mx;
  __syncthreads();
  for (int s = 128; s > 0; s >>= 1) {
    if (t < s) red[t] = fmaxf(red[t], red[t + s]);
    __syncthreads();
  }
  mx = red[0];
  __syncthreads();

  float sum = 0.f;
#pragma unroll
  for (int i = 0; i < 16; i++) {
    xv[i] = __expf(xv[i] - mx);
    sum += xv[i];
  }
  red[t] = sum;
  __syncthreads();
  for (int s = 128; s > 0; s >>= 1) {
    if (t < s) red[t] += red[t + s];
    __syncthreads();
  }
  const float inv = 1.0f / red[0];

  v8bf o0, o1;
#pragma unroll
  for (int i = 0; i < 8; i++) {
    o0[i] = (bf16_t)(xv[i] * inv);
    o1[i] = (bf16_t)(xv[i + 8] * inv);
  }
  *(v8bf*)(o + t * 16)     = o0;
  *(v8bf*)(o + t * 16 + 8) = o1;
}

// ---------------------------------------------------------------------------
// K3: context[d][e] = sum_l ksm[d][l]*v[e][l], per (b,h). grid (8,16), 128 thr.
// Double-buffered async global->LDS DMA of 64-row x 64-l chunks of K_sm and V.
#define CTL  64   // l-chunk (elems)
#define CTLD 72   // padded LDS row stride: 36 dwords -> conflict-free b128 reads
__global__ __launch_bounds__(128) void k_context(
    const bf16_t* __restrict__ ksm, const bf16_t* __restrict__ v,
    bf16_t* __restrict__ ctx) {
  __shared__ bf16_t sk[2][64 * CTLD];   // 9216 B each
  __shared__ bf16_t sv[2][64 * CTLD];
  const int h = blockIdx.x, b = blockIdx.y;
  const int t = threadIdx.x;
  const int wv = t >> 5, lane = t & 31;
  const size_t rbase = (size_t)b * HID + h * DHEAD;
  const bf16_t* kg = ksm + rbase * LDIM;
  const bf16_t* vg = v   + rbase * LDIM;

  const int r  = t >> 1;          // 0..63 : row within tile
  const int cq = (t & 1) * 32;    // 0/32  : 64B half-row (elems)
  const uint32_t ldk0 = (uint32_t)(size_t)&sk[0][r * CTLD + cq];
  const uint32_t ldv0 = (uint32_t)(size_t)&sv[0][r * CTLD + cq];
  const uint32_t ldk1 = (uint32_t)(size_t)&sk[1][r * CTLD + cq];
  const uint32_t ldv1 = (uint32_t)(size_t)&sv[1][r * CTLD + cq];

  // prime chunk 0 (8 async b128 per thread)
  {
    const size_t go = (size_t)r * LDIM + cq;
    async_copy64B(ldk0, kg + go);
    async_copy64B(ldv0, vg + go);
  }

  v8f acc[4] = {};
  for (int c = 0; c < LDIM / CTL; c++) {
    const int p = c & 1;
    if (c + 1 < LDIM / CTL) {
      // issue next chunk into back buffer, then drain current (in-order done)
      const size_t go = (size_t)r * LDIM + (size_t)(c + 1) * CTL + cq;
      async_copy64B(p ? ldk0 : ldk1, kg + go);
      async_copy64B(p ? ldv0 : ldv1, vg + go);
      asm volatile("s_wait_asynccnt 8" ::: "memory");
    } else {
      asm volatile("s_wait_asynccnt 0" ::: "memory");
    }
    __syncthreads();
#pragma unroll
    for (int lof = 0; lof < CTL; lof += 32) {
      v16bf a = frag_a(&sk[p][(wv * 16) * CTLD + lof], CTLD);
#pragma unroll
      for (int ni = 0; ni < 4; ni++) {
        v16bf bb = frag_b(&sv[p][(ni * 16) * CTLD + lof], CTLD);
        acc[ni] = wmma_bf16(a, bb, acc[ni]);
      }
    }
    __syncthreads();
  }

  const int n = lane & 15, half = lane >> 4;
  bf16_t* cp = ctx + (size_t)(b * HEADS + h) * DHEAD * DHEAD;   // [d][e]
#pragma unroll
  for (int ni = 0; ni < 4; ni++)
#pragma unroll
    for (int j = 0; j < 8; j++)
      cp[(wv * 16 + j + 8 * half) * DHEAD + ni * 16 + n] = (bf16_t)acc[ni][j];
}

// ---------------------------------------------------------------------------
// K4: M[b][o][h*64+d] = sum_e Wout[o][h*64+e]*ctx[b,h,d,e]. grid (8,16), 256.
__global__ __launch_bounds__(256) void k_fold_wout(
    const bf16_t* __restrict__ wout, const bf16_t* __restrict__ ctx,
    bf16_t* __restrict__ M) {
  const int h = blockIdx.x, b = blockIdx.y;
  const int wv = threadIdx.x >> 5, lane = threadIdx.x & 31;
  const bf16_t* cp = ctx + (size_t)(b * HEADS + h) * DHEAD * DHEAD;  // [d][e]

  v8f acc[2][4] = {};
#pragma unroll
  for (int e0 = 0; e0 < DHEAD; e0 += 32) {
    v16bf bfr[4];
#pragma unroll
    for (int ni = 0; ni < 4; ni++)
      bfr[ni] = frag_b(cp + (ni * 16) * DHEAD + e0, DHEAD);
#pragma unroll
    for (int mi = 0; mi < 2; mi++) {
      v16bf a = frag_a(wout + (size_t)(wv * 32 + mi * 16) * HID + h * DHEAD + e0, HID);
#pragma unroll
      for (int ni = 0; ni < 4; ni++)
        acc[mi][ni] = wmma_bf16(a, bfr[ni], acc[mi][ni]);
    }
  }
  const int n = lane & 15, half = lane >> 4;
#pragma unroll
  for (int mi = 0; mi < 2; mi++)
#pragma unroll
    for (int ni = 0; ni < 4; ni++)
#pragma unroll
      for (int j = 0; j < 8; j++)
        M[((size_t)b * CIN + wv * 32 + mi * 16 + j + 8 * half) * HID +
          h * DHEAD + ni * 16 + n] = (bf16_t)acc[mi][ni][j];
}

// ---------------------------------------------------------------------------
// K5: final = M(256x512) @ Q(512x4096) + b_out. Q is stored transposed so B
// fragments are contiguous -> no LDS. grid (64 l, 4 o, 16 b), 128 threads.
__global__ __launch_bounds__(128) void k_out_gemm(
    const bf16_t* __restrict__ M, const bf16_t* __restrict__ qt,
    const float* __restrict__ bo, float* __restrict__ out) {
  const int b = blockIdx.z, ot = blockIdx.y;
  const int l0 = blockIdx.x * 64;
  const int wv = threadIdx.x >> 5, lane = threadIdx.x & 31;
  const int mq = wv >> 1, nq = wv & 1;
  const int o0 = ot * 64 + mq * 32;

  const bf16_t* Mb = M + (size_t)b * CIN * HID;
  const bf16_t* qb = qt + (size_t)b * LDIM * HID;

  v8f acc[2][2] = {};
  for (int k0 = 0; k0 < HID; k0 += 32) {
    v16bf af[2], bfr[2];
#pragma unroll
    for (int mi = 0; mi < 2; mi++)
      af[mi] = frag_a(Mb + (size_t)(o0 + mi * 16) * HID + k0, HID);
#pragma unroll
    for (int ni = 0; ni < 2; ni++)
      bfr[ni] = frag_b(qb + (size_t)(l0 + nq * 32 + ni * 16) * HID + k0, HID);
#pragma unroll
    for (int mi = 0; mi < 2; mi++)
#pragma unroll
      for (int ni = 0; ni < 2; ni++)
        acc[mi][ni] = wmma_bf16(af[mi], bfr[ni], acc[mi][ni]);
  }

  const int n = lane & 15, half = lane >> 4;
#pragma unroll
  for (int mi = 0; mi < 2; mi++)
#pragma unroll
    for (int ni = 0; ni < 2; ni++)
#pragma unroll
      for (int j = 0; j < 8; j++) {
        const int o = o0 + mi * 16 + j + 8 * half;
        out[((size_t)b * CIN + o) * LDIM + l0 + nq * 32 + ni * 16 + n] =
            acc[mi][ni][j] + bo[o];
      }
}

// ---------------------------------------------------------------------------
extern "C" void kernel_launch(void* const* d_in, const int* in_sizes, int n_in,
                              void* d_out, int out_size, void* d_ws, size_t ws_size,
                              hipStream_t stream) {
  const float* x    = (const float*)d_in[0];   // (16,256,4096)
  const float* wqkv = (const float*)d_in[1];   // (1536,256)
  const float* wout = (const float*)d_in[2];   // (256,512)
  const float* bo   = (const float*)d_in[3];   // (256,)
  float* out = (float*)d_out;                  // (16,256,4096)

  char* ws = (char*)d_ws;
  bf16_t* wqkv_bf = (bf16_t*)(ws + OFF_WQKV);
  bf16_t* wout_bf = (bf16_t*)(ws + OFF_WOUT);
  bf16_t* qt      = (bf16_t*)(ws + OFF_QT);
  float*  kbuf    = (float* )(ws + OFF_K);
  bf16_t* ksm     = (bf16_t*)(ws + OFF_KSM);
  bf16_t* vbuf    = (bf16_t*)(ws + OFF_V);
  bf16_t* ctx     = (bf16_t*)(ws + OFF_CTX);
  bf16_t* Mb      = (bf16_t*)(ws + OFF_M);

  k_convert_weights<<<dim3((OQKV * CIN + 255) / 256), dim3(256), 0, stream>>>(
      wqkv, wout, wqkv_bf, wout_bf);

  k_qkv_gemm<<<dim3(LDIM / 64, OQKV / 64, BATCH), dim3(128), 0, stream>>>(
      x, wqkv_bf, qt, kbuf, vbuf);

  k_softmax<<<dim3(BATCH * HID), dim3(256), 0, stream>>>(kbuf, ksm);

  k_context<<<dim3(HEADS, BATCH), dim3(128), 0, stream>>>(ksm, vbuf, ctx);

  k_fold_wout<<<dim3(HEADS, BATCH), dim3(256), 0, stream>>>(wout_bf, ctx, Mb);

  k_out_gemm<<<dim3(LDIM / 64, CIN / 64, BATCH), dim3(128), 0, stream>>>(
      Mb, qt, bo, out);
}
